// slot_dec_86303072846146
// MI455X (gfx1250) — matrix operand order repaired and checked by reading, not compiled, not run on hardware
//
#include <hip/hip_runtime.h>

// Problem dimensions (from reference)
#define H      1024
#define BATCH  32
#define SEQ    128
#define LABEL  122
#define S_TOT  (SEQ * BATCH)   // 4096 flattened scan steps
#define K4     4096            // 4*H  (gate rows / xc width)
#define K5     5120            // 5*H  (W_ih width)
#define NWG_SCAN 64

typedef __attribute__((ext_vector_type(16))) __bf16        v16bf;
typedef __attribute__((ext_vector_type(8)))  float         v8f;
typedef __attribute__((ext_vector_type(8)))  unsigned int  v8u;

// ---------------------------------------------------------------- helpers ---
__device__ __forceinline__ unsigned short f2bf(float f) {
    unsigned int u = __builtin_bit_cast(unsigned int, f);
    unsigned int r = u + 0x7FFFu + ((u >> 16) & 1u);   // round-to-nearest-even
    return (unsigned short)(r >> 16);
}
__device__ __forceinline__ unsigned int rotl32(unsigned int x, int r) {
    return (x << r) | (x >> (32 - r));
}
// Exact JAX threefry2x32 with key = jax.random.key(42) -> (0, 42)
__device__ __forceinline__ void threefry2x32_42(unsigned int c0, unsigned int c1,
                                                unsigned int& o0, unsigned int& o1) {
    const unsigned int ks0 = 0u, ks1 = 42u;
    const unsigned int ks2 = 0x1BD11BDAu ^ ks0 ^ ks1;
    unsigned int x0 = c0 + ks0, x1 = c1 + ks1;
#define TF_R(r) { x0 += x1; x1 = rotl32(x1, r); x1 ^= x0; }
    TF_R(13) TF_R(15) TF_R(26) TF_R(6)   x0 += ks1; x1 += ks2 + 1u;
    TF_R(17) TF_R(29) TF_R(16) TF_R(24)  x0 += ks2; x1 += ks0 + 2u;
    TF_R(13) TF_R(15) TF_R(26) TF_R(6)   x0 += ks0; x1 += ks1 + 3u;
    TF_R(17) TF_R(29) TF_R(16) TF_R(24)  x0 += ks1; x1 += ks2 + 4u;
    TF_R(13) TF_R(15) TF_R(26) TF_R(6)   x0 += ks2; x1 += ks0 + 5u;
#undef TF_R
    o0 = x0; o1 = x1;
}
__device__ __forceinline__ float bits2unif(unsigned int b) {
    return __builtin_bit_cast(float, (b >> 9) | 0x3F800000u) - 1.0f;
}
__device__ __forceinline__ float sigf(float x) { return 1.0f / (1.0f + __expf(-x)); }

// ---------------------------------------------------- kernel 0: init state ---
__global__ __launch_bounds__(256) void k_init(float* __restrict__ bias,
                                              const float* __restrict__ b_ih,
                                              const float* __restrict__ b_hh,
                                              float* __restrict__ hbuf,
                                              int* __restrict__ bar) {
    int i = blockIdx.x * 256 + threadIdx.x;
    if (i < K4) bias[i] = b_ih[i] + b_hh[i];
    if (i < H)  hbuf[i] = 0.0f;
    if (i < 2)  bar[i]  = 0;
}

// -------------------------- kernel 1: dropout + concat + transpose -> bf16 ---
__device__ __forceinline__ void emit_xc(const float* __restrict__ x,
                                        const float* __restrict__ hi,
                                        unsigned short* __restrict__ XC,
                                        unsigned int e, unsigned int bits) {
    unsigned int m = e >> 12;            // row = t*32 + b
    unsigned int k = e & 4095u;
    unsigned int b = m & 31u, t = m >> 5;
    size_t base = ((size_t)b * SEQ + t) * (size_t)(2 * H);
    float val = (k < 2048u) ? x[base + k] : hi[base + (k - 2048u)];
    float keep = (bits2unif(bits) < 0.9f) ? (1.0f / 0.9f) : 0.0f;
    XC[e] = f2bf(val * keep);
}

__global__ __launch_bounds__(256) void k_dropout_concat(const float* __restrict__ x,
                                                        const float* __restrict__ hi,
                                                        unsigned short* __restrict__ XC) {
    const unsigned int M = (unsigned int)S_TOT * (unsigned int)K4 / 2u;  // 8388608
    unsigned int i = blockIdx.x * 256u + threadIdx.x;
    if (i >= M) return;
    unsigned int o0, o1;
    threefry2x32_42(i, i + M, o0, o1);
    emit_xc(x, hi, XC, i,     o0);
    emit_xc(x, hi, XC, i + M, o1);
}

// ----------------- kernel 1b: W_ih[:, :4H] -> bf16 copy (one-time stream) ---
__global__ __launch_bounds__(256) void k_wconv(const float* __restrict__ W_ih,
                                               unsigned short* __restrict__ WB) {
    unsigned int i = blockIdx.x * 256u + threadIdx.x;   // 16,777,216 elems
    unsigned int n = i >> 12, k = i & 4095u;
    WB[i] = f2bf(W_ih[(size_t)n * K5 + k]);
}

// -------- kernel 2: G_x = XCbf @ WB.T  (bf16 WMMA, async-LDS double buffer) --
// 128x128 tile/WG, 8 waves (2x4), each wave 64x32 = 8 wmma accumulators.
// Tiles staged with GLOBAL_LOAD_ASYNC_TO_LDS_B128 (ASYNCcnt), DMA scatters rows
// to a padded stride-20-uint layout so fragment ds_load_b128s stay bank-spread.
__global__ __launch_bounds__(256) void k_gemm_gx(const unsigned short* __restrict__ XCbf,
                                                 const unsigned short* __restrict__ WBbf,
                                                 float* __restrict__ Gx) {
    __shared__ unsigned int Alds[2][128][20];
    __shared__ unsigned int Blds[2][128][20];

    const int tid   = threadIdx.x;
    const int lane  = tid & 31;          // wave32 (gfx1250)
    const int wid   = tid >> 5;
    const int wr    = wid >> 2;          // 0..1  (M)
    const int wc    = wid & 3;           // 0..3  (N)
    const int m0    = blockIdx.y * 128;
    const int n0    = blockIdx.x * 128;
    const int lrow  = lane & 15;
    const int lhalf = lane >> 4;

    v8f acc[4][2];
    #pragma unroll
    for (int a = 0; a < 4; ++a)
        #pragma unroll
        for (int b = 0; b < 2; ++b) acc[a][b] = (v8f){};

    // issue one 128x32 bf16 A-tile + B-tile: 4 async b128 per wave-thread
    auto issue = [&](int buf, int kb) {
        #pragma unroll
        for (int j = 0; j < 2; ++j) {
            int id  = tid + j * 256;          // 0..511 transfers of 16B
            int row = id >> 2;
            int seg = id & 3;
            const char* ga = (const char*)XCbf +
                (((size_t)(m0 + row) * K4 + kb) << 1) + seg * 16;
            unsigned la = (unsigned)(uintptr_t)&Alds[buf][row][seg * 4];
            asm volatile("global_load_async_to_lds_b128 %0, %1, off"
                         :: "v"(la), "v"(ga) : "memory");
            const char* gb = (const char*)WBbf +
                (((size_t)(n0 + row) * K4 + kb) << 1) + seg * 16;
            unsigned lb = (unsigned)(uintptr_t)&Blds[buf][row][seg * 4];
            asm volatile("global_load_async_to_lds_b128 %0, %1, off"
                         :: "v"(lb), "v"(gb) : "memory");
        }
    };

    issue(0, 0);
    for (int kb = 0; kb < K4; kb += 32) {
        const int cur = (kb >> 5) & 1;
        if (kb + 32 < K4) {
            issue(cur ^ 1, kb + 32);
            // oldest 4 (current tile) complete; newest 4 still in flight
            asm volatile("s_wait_asynccnt 0x4" ::: "memory");
        } else {
            asm volatile("s_wait_asynccnt 0x0" ::: "memory");
        }
        __syncthreads();

        v16bf afrag[4], bfrag[2];
        #pragma unroll
        for (int mf = 0; mf < 4; ++mf) {           // A 16x32 bf16 lane layout
            int r = wr * 64 + mf * 16 + lrow;
            v8u raw;
            #pragma unroll
            for (int v = 0; v < 8; ++v) {
                int col = (v < 4) ? (lhalf * 4 + v) : (4 + lhalf * 4 + v);
                raw[v] = Alds[cur][r][col];
            }
            afrag[mf] = __builtin_bit_cast(v16bf, raw);
        }
        #pragma unroll
        for (int nf = 0; nf < 2; ++nf) {           // B 32x16 bf16 lane layout
            int r = wc * 32 + nf * 16 + lrow;
            v8u raw;
            #pragma unroll
            for (int v = 0; v < 8; ++v) raw[v] = Blds[cur][r][(v < 4) ? (lhalf * 4 + v) : (4 + lhalf * 4 + v)];
            bfrag[nf] = __builtin_bit_cast(v16bf, raw);
        }
        #pragma unroll
        for (int mf = 0; mf < 4; ++mf)
            #pragma unroll
            for (int nf = 0; nf < 2; ++nf)
                acc[mf][nf] = __builtin_amdgcn_wmma_f32_16x16x32_bf16(
                    false, afrag[mf], false, bfrag[nf], (short)0, acc[mf][nf], false, false);
        __syncthreads();
    }

    // epilogue: C/D 16x16 f32 layout -> Gx
    #pragma unroll
    for (int mf = 0; mf < 4; ++mf)
        #pragma unroll
        for (int nf = 0; nf < 2; ++nf)
            #pragma unroll
            for (int r = 0; r < 8; ++r) {
                int Mi = m0 + wr * 64 + mf * 16 + lhalf * 8 + r;
                int Ni = n0 + wc * 32 + nf * 16 + lrow;
                Gx[(size_t)Mi * K4 + Ni] = acc[mf][nf][r];
            }
}

// ----------- kernel 3: persistent sequential LSTM scan (4096 steps) ---------
__global__ __launch_bounds__(256) void k_scan(const float* __restrict__ Gx,
                                              const float* __restrict__ W_ih,
                                              const float* __restrict__ W_hh,
                                              const float* __restrict__ bias,
                                              float* __restrict__ hbuf,
                                              float* __restrict__ allout,
                                              int* __restrict__ bar) {
    __shared__ float vec[2 * H];    // [out_prev(1024) ; h(1024)]
    __shared__ float part[64][4];
    __shared__ float gvs[64];

    const int tid  = threadIdx.x;
    const int j0   = blockIdx.x * (H / NWG_SCAN);  // 16 hidden units per WG
    const int row  = tid >> 2;                     // 0..63 local gate-rows
    const int q    = tid & 3;                      // quarter of the 2048-dot
    const int gate = row >> 4;
    const int jj   = row & 15;
    const int grow = gate * H + j0 + jj;           // global gate row
    const float* wih = W_ih + (size_t)grow * K5 + K4;  // out_prev block
    const float* whh = W_hh + (size_t)grow * H;
    int* cnt = bar;
    int* gen = bar + 1;
    float c_reg = 0.0f;                            // cell state, register-resident
    const unsigned vbase = (unsigned)(uintptr_t)&vec[0];

    for (int s = 0; s < S_TOT; ++s) {
        // [out_prev ; h] gather is a pure global->LDS copy: use async DMA path
        if (s >= BATCH) {
            const char* ga = (const char*)(allout + (size_t)(s - BATCH) * H) + tid * 16;
            unsigned la = vbase + tid * 16;
            asm volatile("global_load_async_to_lds_b128 %0, %1, off"
                         :: "v"(la), "v"(ga) : "memory");
        } else {
            *(float4*)&vec[tid * 4] = make_float4(0.f, 0.f, 0.f, 0.f);
        }
        {
            const char* gh = (const char*)hbuf + tid * 16;
            unsigned lh = vbase + 4096 + tid * 16;
            asm volatile("global_load_async_to_lds_b128 %0, %1, off"
                         :: "v"(lh), "v"(gh) : "memory");
        }
        asm volatile("s_wait_asynccnt 0x0" ::: "memory");
        __syncthreads();

        float acc = 0.0f;
        #pragma unroll 8
        for (int u = 0; u < 64; ++u) {             // W_ih[:,4H:] @ out_prev
            int c = u * 16 + q * 4;
            float4 w = *(const float4*)(wih + c);
            acc += w.x * vec[c] + w.y * vec[c + 1] + w.z * vec[c + 2] + w.w * vec[c + 3];
        }
        #pragma unroll 8
        for (int u = 0; u < 64; ++u) {             // W_hh @ h
            int c = u * 16 + q * 4;
            float4 w = *(const float4*)(whh + c);
            const float* v = vec + H + c;
            acc += w.x * v[0] + w.y * v[1] + w.z * v[2] + w.w * v[3];
        }
        part[row][q] = acc;
        __syncthreads();

        if (tid < 64) {
            int g2 = (tid >> 4) * H + j0 + (tid & 15);
            gvs[tid] = Gx[(size_t)s * K4 + g2] + bias[g2]
                     + part[tid][0] + part[tid][1] + part[tid][2] + part[tid][3];
        }
        __syncthreads();

        if (tid < 16) {
            int j = j0 + tid;
            float ig = sigf(gvs[tid]);
            float fg = sigf(gvs[16 + tid]);
            float gg = tanhf(gvs[32 + tid]);
            float og = sigf(gvs[48 + tid]);
            c_reg = fg * c_reg + ig * gg;
            float h = og * tanhf(c_reg);
            hbuf[j] = h;
            allout[(size_t)s * H + j] = h;
        }
        __threadfence();
        __syncthreads();

        // device-wide monotonic barrier (init kernel zeroes cnt/gen each launch)
        if (tid == 0) {
            int prev = atomicAdd(cnt, 1);
            if (prev == NWG_SCAN * (s + 1) - 1) {
                atomicExch(gen, s + 1);
            } else {
                while (atomicAdd(gen, 0) < s + 1) __builtin_amdgcn_s_sleep(1);
            }
        }
        __syncthreads();
        __threadfence();
    }
}

// ------------------- kernel 4: output projection (B,L,H)@(LABEL,H)^T --------
__global__ __launch_bounds__(128) void k_proj(const float* __restrict__ allout,
                                              const float* __restrict__ W_fc,
                                              const float* __restrict__ b_fc,
                                              float* __restrict__ out) {
    __shared__ float hrow[H];
    int bt = blockIdx.x;                 // b*SEQ + t
    int b = bt / SEQ, t = bt - b * SEQ;
    int s = t * BATCH + b;               // scan-order row
    for (int i = threadIdx.x; i < H; i += 128) hrow[i] = allout[(size_t)s * H + i];
    __syncthreads();
    int l = threadIdx.x;
    if (l < LABEL) {
        const float* w = W_fc + (size_t)l * H;
        float acc = 0.0f;
        for (int k = 0; k < H; k += 4) {
            float4 wv = *(const float4*)(w + k);
            acc += wv.x * hrow[k] + wv.y * hrow[k + 1] + wv.z * hrow[k + 2] + wv.w * hrow[k + 3];
        }
        out[(size_t)bt * LABEL + l] = acc + b_fc[l];
    }
}

// ---------------------------------------------------------------- launcher --
extern "C" void kernel_launch(void* const* d_in, const int* in_sizes, int n_in,
                              void* d_out, int out_size, void* d_ws, size_t ws_size,
                              hipStream_t stream) {
    const float* x    = (const float*)d_in[0];
    const float* hi   = (const float*)d_in[1];
    const float* W_ih = (const float*)d_in[2];
    const float* W_hh = (const float*)d_in[3];
    const float* b_ih = (const float*)d_in[4];
    const float* b_hh = (const float*)d_in[5];
    const float* W_fc = (const float*)d_in[6];
    const float* b_fc = (const float*)d_in[7];
    float* out = (float*)d_out;

    char* ws = (char*)d_ws;
    size_t off = 0;
    auto take = [&](size_t bytes) {
        size_t o = off;
        off += (bytes + 255) & ~(size_t)255;
        return o;
    };
    unsigned short* XC  = (unsigned short*)(ws + take((size_t)S_TOT * K4 * 2)); // 33.5 MB
    unsigned short* WB  = (unsigned short*)(ws + take((size_t)K4 * K4 * 2));    // 33.5 MB
    float* Gx           = (float*)(ws + take((size_t)S_TOT * K4 * 4));          // 67 MB
    float* allout       = (float*)(ws + take((size_t)S_TOT * H * 4));           // 16.8 MB
    float* hbuf         = (float*)(ws + take((size_t)H * 4));
    float* bias         = (float*)(ws + take((size_t)K4 * 4));
    int*   bar          = (int*)(ws + take(256));
    (void)ws_size; (void)in_sizes; (void)n_in; (void)out_size;

    k_init<<<16, 256, 0, stream>>>(bias, b_ih, b_hh, hbuf, bar);
    k_dropout_concat<<<(S_TOT * K4 / 2) / 256, 256, 0, stream>>>(x, hi, XC);
    k_wconv<<<(K4 * K4) / 256, 256, 0, stream>>>(W_ih, WB);
    dim3 gg(K4 / 128, S_TOT / 128);
    k_gemm_gx<<<gg, 256, 0, stream>>>(XC, WB, Gx);
    k_scan<<<NWG_SCAN, 256, 0, stream>>>(Gx, W_ih, W_hh, bias, hbuf, allout, bar);
    k_proj<<<BATCH * SEQ, 128, 0, stream>>>(allout, W_fc, b_fc, out);
}
